// QuantizedECGNet_62088047231423
// MI455X (gfx1250) — matrix operand order repaired and checked by reading, not compiled, not run on hardware
//
#include <hip/hip_runtime.h>
#include <hip/hip_bf16.h>

typedef __attribute__((ext_vector_type(16))) _Float16 v16h;
typedef __attribute__((ext_vector_type(8)))  float    v8f;

__device__ __forceinline__ float clampf(float v, float lo, float hi) {
  return fminf(fmaxf(v, lo), hi);
}

// ---------------------------------------------------------------------------
// Weight scale: one block per tensor, max-abs reduction -> scale = max/n
// ---------------------------------------------------------------------------
__global__ void k_wscale(const float* w0, const float* w1, const float* w2,
                         const float* w3, const float* w4, const float* w5,
                         const float* w6, const float* w7, const float* w8,
                         float* __restrict__ wsc) {
  __shared__ float red[256];
  const float* ptrs[9] = {w0, w1, w2, w3, w4, w5, w6, w7, w8};
  const int    sz[9]   = {128, 1152, 3456, 14336, 20480, 12288, 13824, 13824, 1088};
  const float  nv[9]   = {7.f, 7.f, 7.f, 7.f, 7.f, 7.f, 7.f, 7.f, 127.f};
  const int t = blockIdx.x;
  const float* p = ptrs[t];
  float m = 0.f;
  for (int i = threadIdx.x; i < sz[t]; i += 256) m = fmaxf(m, fabsf(p[i]));
  red[threadIdx.x] = m;
  __syncthreads();
  for (int s = 128; s > 0; s >>= 1) {
    if ((int)threadIdx.x < s) red[threadIdx.x] = fmaxf(red[threadIdx.x], red[threadIdx.x + s]);
    __syncthreads();
  }
  if (threadIdx.x == 0) wsc[t] = red[0] / nv[t];
}

// Quantize weights to integer codes stored as f16
__global__ void k_wquant(const float* __restrict__ w, const float* __restrict__ sc,
                         float lo, float hi, _Float16* __restrict__ out, int n) {
  int i = blockIdx.x * blockDim.x + threadIdx.x;
  if (i >= n) return;
  out[i] = (_Float16)clampf(rintf(w[i] / sc[0]), lo, hi);
}

// Input QuantIdentity: signed 8-bit codes
__global__ void k_inq(const float* __restrict__ x, const float* __restrict__ s,
                      _Float16* __restrict__ out, int n) {
  int i = blockIdx.x * blockDim.x + threadIdx.x;
  if (i >= n) return;
  out[i] = (_Float16)clampf(rintf(x[i] / s[0]), -128.f, 127.f);
}

// ---------------------------------------------------------------------------
// Direct conv (small-channel layers 1..3), codes in -> 4-bit relu codes out
// ---------------------------------------------------------------------------
template<int C, int K, int O, int Lin, int Lout, int STRIDE, int PAD>
__global__ void k_conv_direct(const _Float16* __restrict__ in, const _Float16* __restrict__ wq,
                              const float* __restrict__ bias,
                              const float* __restrict__ s, int is_idx,
                              const float* __restrict__ wsc, int ws_idx,
                              _Float16* __restrict__ out) {
  constexpr int TOTAL = 2048 * O * Lout;
  int idx = blockIdx.x * blockDim.x + threadIdx.x;
  if (idx >= TOTAL) return;
  const int l = idx % Lout;
  const int o = (idx / Lout) % O;
  const int b = idx / (Lout * O);
  const _Float16* ip = in + (size_t)b * C * Lin;
  const _Float16* wp = wq + o * C * K;
  float acc = 0.f;
  const int basex = l * STRIDE - PAD;
  for (int c = 0; c < C; ++c) {
#pragma unroll
    for (int k = 0; k < K; ++k) {
      const int xx = basex + k;
      if (xx >= 0 && xx < Lin)
        acc += (float)ip[c * Lin + xx] * (float)wp[c * K + k];
    }
  }
  const float comb = s[is_idx] * wsc[ws_idx];
  const float bq = rintf(bias[o] / comb);
  const float v = (acc + bq) * comb;
  out[idx] = (_Float16)clampf(rintf(v / s[is_idx + 1]), 0.f, 15.f);
}

// Maxpool on codes (scales positive -> max commutes)
template<int CH, int Lin, int Lout, int PK, int PS>
__global__ void k_pool(const _Float16* __restrict__ in, _Float16* __restrict__ out) {
  constexpr int TOTAL = 2048 * CH * Lout;
  int idx = blockIdx.x * blockDim.x + threadIdx.x;
  if (idx >= TOTAL) return;
  const int l = idx % Lout;
  const int rest = idx / Lout;  // b*CH + c
  const _Float16* ip = in + (size_t)rest * Lin + l * PS;
  float m = (float)ip[0];
#pragma unroll
  for (int k = 1; k < PK; ++k) m = fmaxf(m, (float)ip[k]);
  out[idx] = (_Float16)m;
}

// ---------------------------------------------------------------------------
// WMMA implicit-im2col conv (layers 4..7): one block per batch, LDS-staged
// input row + weights, each wave owns a 16x16 (l x o) tile; K-dim = C*K.
// A/B fragment gathers follow CDNA5 ISA 7.12.2 16-bit layouts.
// ---------------------------------------------------------------------------
template<int C, int K, int O, int Npad, int Lin, int Lout, int PAD>
__global__ void __launch_bounds__(256)
k_conv_wmma(const _Float16* __restrict__ in, const _Float16* __restrict__ wq,
            const float* __restrict__ bias,
            const float* __restrict__ s, int is_idx,
            const float* __restrict__ wsc, int ws_idx,
            _Float16* __restrict__ out) {
  constexpr int CK = C * K;
  static_assert(CK % 32 == 0, "GEMM K-dim must be multiple of 32");
  constexpr int NCH = CK / 32;
  constexpr int Wp  = Lin + 2 * PAD + 32;   // slack covers tail-row overrun
  constexpr int MT  = (Lout + 15) / 16;
  constexpr int NT  = Npad / 16;
  __shared__ _Float16 s_in[C * Wp];
  __shared__ _Float16 s_w[Npad * CK];

  const int b   = blockIdx.x;
  const int tid = threadIdx.x;

  for (int i = tid; i < C * Wp; i += 256) {           // zero-padded input row
    const int c = i / Wp, xx = i % Wp;
    const int l = xx - PAD;
    _Float16 v = (_Float16)0.f;
    if (l >= 0 && l < Lin) v = in[((size_t)b * C + c) * Lin + l];
    s_in[i] = v;
  }
  for (int i = tid; i < Npad * CK; i += 256) {        // zero-padded weights
    const int o = i / CK;
    s_w[i] = (o < O) ? wq[i] : (_Float16)0.f;
  }
  __syncthreads();

  const float comb = s[is_idx] * wsc[ws_idx];
  const float sout = s[is_idx + 1];

  const int wave = tid >> 5;
  const int lane = tid & 31;
  const int m16  = lane & 15;
  const int hi   = lane >> 4;

  for (int tile = wave; tile < MT * NT; tile += 8) {
    const int mt = tile / NT;
    const int nt = tile % NT;
    const int lbase = mt * 16 + m16;                  // this lane's A row (output l)
    v8f acc = {};
#pragma unroll
    for (int ch = 0; ch < NCH; ++ch) {
      v16h a;
#pragma unroll
      for (int j = 0; j < 8; ++j) {
        const int qb = ch * 32 + ((j < 4) ? (8 * hi + 2 * j)
                                          : (16 + 8 * hi + 2 * (j - 4)));
#pragma unroll
        for (int t2 = 0; t2 < 2; ++t2) {
          const int q = qb + t2;                      // GEMM-K index = c*K + k
          const int c = q / K;
          const int k = q % K;
          a[2 * j + t2] = s_in[c * Wp + lbase + k];
        }
      }
      v16h bf;                                        // lane = col n, K in two 8-runs (8*hi, 16+8*hi)
      const _Float16* bp = &s_w[(nt * 16 + m16) * CK + ch * 32 + 8 * hi];
#pragma unroll
      for (int hh = 0; hh < 16; ++hh) bf[hh] = bp[hh < 8 ? hh : hh + 8];
      acc = __builtin_amdgcn_wmma_f32_16x16x32_f16(false, a, false, bf,
                                                   (short)0, acc, false, false);
    }
#pragma unroll
    for (int r = 0; r < 8; ++r) {                     // C/D layout: M = r + 8*hi
      const int m = mt * 16 + r + 8 * hi;
      const int o = nt * 16 + m16;
      if (m < Lout && o < O) {
        const float bq = rintf(bias[o] / comb);
        const float v  = (acc[r] + bq) * comb;
        out[((size_t)b * O + o) * Lout + m] =
            (_Float16)clampf(rintf(v / sout), 0.f, 15.f);
      }
    }
  }
}

// ---------------------------------------------------------------------------
// Linear1 (2048x216 @ 216x64) via WMMA, K padded 216->224
// ---------------------------------------------------------------------------
__global__ void __launch_bounds__(128)
k_lin1_wmma(const _Float16* __restrict__ flat, const _Float16* __restrict__ wq,
            const float* __restrict__ bias, const float* __restrict__ s,
            const float* __restrict__ wsc, _Float16* __restrict__ out) {
  constexpr int KG = 216, KP = 224, N = 64, NCH = KP / 32;
  __shared__ _Float16 s_a[16 * KP];
  __shared__ _Float16 s_b[N * KP];
  const int b0  = blockIdx.x * 16;
  const int tid = threadIdx.x;
  for (int i = tid; i < 16 * KP; i += 128) {
    const int m = i / KP, q = i % KP;
    s_a[i] = (q < KG) ? flat[(size_t)(b0 + m) * KG + q] : (_Float16)0.f;
  }
  for (int i = tid; i < N * KP; i += 128) {
    const int n2 = i / KP, q = i % KP;
    s_b[i] = (q < KG) ? wq[n2 * KG + q] : (_Float16)0.f;
  }
  __syncthreads();
  const int wave = tid >> 5, lane = tid & 31, m16 = lane & 15, hi = lane >> 4;
  const int nt = wave;  // 4 waves, 4 N-tiles
  v8f acc = {};
#pragma unroll
  for (int ch = 0; ch < NCH; ++ch) {
    v16h a;
#pragma unroll
    for (int j = 0; j < 8; ++j) {
      const int qb = ch * 32 + ((j < 4) ? (8 * hi + 2 * j) : (16 + 8 * hi + 2 * (j - 4)));
      a[2 * j]     = s_a[m16 * KP + qb];
      a[2 * j + 1] = s_a[m16 * KP + qb + 1];
    }
    v16h bf;
    const _Float16* bp = &s_b[(nt * 16 + m16) * KP + ch * 32 + 8 * hi];
#pragma unroll
    for (int hh = 0; hh < 16; ++hh) bf[hh] = bp[hh < 8 ? hh : hh + 8];
    acc = __builtin_amdgcn_wmma_f32_16x16x32_f16(false, a, false, bf,
                                                 (short)0, acc, false, false);
  }
  const float comb = s[7] * wsc[7];
  const float sout = s[8];
#pragma unroll
  for (int r = 0; r < 8; ++r) {
    const int m  = r + 8 * hi;
    const int n2 = nt * 16 + m16;
    const float bq = rintf(bias[n2] / comb);
    const float v  = (acc[r] + bq) * comb;
    out[(size_t)(b0 + m) * N + n2] = (_Float16)clampf(rintf(v / sout), 0.f, 15.f);
  }
}

// Linear2 (64 -> 17), final f32 output, no relu/requant
__global__ void k_lin2(const _Float16* __restrict__ h, const _Float16* __restrict__ wq,
                       const float* __restrict__ bias, const float* __restrict__ s,
                       const float* __restrict__ wsc, float* __restrict__ out) {
  int idx = blockIdx.x * blockDim.x + threadIdx.x;
  if (idx >= 2048 * 17) return;
  const int n = idx % 17, b = idx / 17;
  const _Float16* hp = h + (size_t)b * 64;
  const _Float16* wp = wq + n * 64;
  float acc = 0.f;
#pragma unroll
  for (int k = 0; k < 64; ++k) acc += (float)hp[k] * (float)wp[k];
  const float comb = s[8] * wsc[8];
  const float bq = clampf(rintf(bias[n] / comb), -2147483648.f, 2147483647.f);
  out[idx] = (acc + bq) * comb;
}

// ---------------------------------------------------------------------------
extern "C" void kernel_launch(void* const* d_in, const int* in_sizes, int n_in,
                              void* d_out, int out_size, void* d_ws, size_t ws_size,
                              hipStream_t stream) {
  (void)in_sizes; (void)n_in; (void)out_size; (void)ws_size;
  const float* x = (const float*)d_in[0];
  const float* W[9]  = {(const float*)d_in[1],  (const float*)d_in[3],  (const float*)d_in[5],
                        (const float*)d_in[7],  (const float*)d_in[9],  (const float*)d_in[11],
                        (const float*)d_in[13], (const float*)d_in[15], (const float*)d_in[17]};
  const float* Bv[9] = {(const float*)d_in[2],  (const float*)d_in[4],  (const float*)d_in[6],
                        (const float*)d_in[8],  (const float*)d_in[10], (const float*)d_in[12],
                        (const float*)d_in[14], (const float*)d_in[16], (const float*)d_in[18]};
  const float* s = (const float*)d_in[19];
  float* outp = (float*)d_out;

  char* base = (char*)d_ws;
  float* wsc = (float*)base;   // 9 weight scales
  size_t off = 256;
  auto allocH = [&](size_t nh) {
    _Float16* p = (_Float16*)(base + off);
    off = (off + nh * sizeof(_Float16) + 255) & ~(size_t)255;
    return p;
  };
  const int wsz[9] = {128, 1152, 3456, 14336, 20480, 12288, 13824, 13824, 1088};
  _Float16* wq[9];
  for (int t = 0; t < 9; ++t) wq[t] = allocH((size_t)wsz[t]);
  const size_t REG = 2048ull * 8 * 1800;   // largest intermediate (conv1 out)
  _Float16* RA = allocH(REG);              // ping-pong regions
  _Float16* RB = allocH(REG);

  // on-device weight quantization
  k_wscale<<<9, 256, 0, stream>>>(W[0], W[1], W[2], W[3], W[4], W[5], W[6], W[7], W[8], wsc);
  for (int t = 0; t < 9; ++t) {
    const float lo = (t == 8) ? -128.f : -8.f;
    const float hi = (t == 8) ?  127.f :  7.f;
    k_wquant<<<(wsz[t] + 255) / 256, 256, 0, stream>>>(W[t], wsc + t, lo, hi, wq[t], wsz[t]);
  }
  { int n = 2048 * 3600; k_inq<<<(n + 255) / 256, 256, 0, stream>>>(x, s, RA, n); }

  // conv1..3: direct VALU (tiny channel counts, memory-shaped)
  { int n = 2048*8*1800;  k_conv_direct<1,16,8,3600,1800,2,7><<<(n+255)/256,256,0,stream>>>(RA, wq[0], Bv[0], s, 0, wsc, 0, RB); }
  { int n = 2048*8*449;   k_pool<8,1800,449,8,4><<<(n+255)/256,256,0,stream>>>(RB, RA); }
  { int n = 2048*12*224;  k_conv_direct<8,12,12,449,224,2,5><<<(n+255)/256,256,0,stream>>>(RA, wq[1], Bv[1], s, 1, wsc, 1, RB); }
  { int n = 2048*12*111;  k_pool<12,224,111,4,2><<<(n+255)/256,256,0,stream>>>(RB, RA); }
  { int n = 2048*32*111;  k_conv_direct<12,9,32,111,111,1,4><<<(n+255)/256,256,0,stream>>>(RA, wq[2], Bv[2], s, 2, wsc, 2, RB); }
  { int n = 2048*32*54;   k_pool<32,111,54,5,2><<<(n+255)/256,256,0,stream>>>(RB, RA); }

  // conv4..7: WMMA implicit-im2col GEMM (one block per batch sample)
  k_conv_wmma<32,7,64,64,54,54,3><<<2048,256,0,stream>>>(RA, wq[3], Bv[3], s, 3, wsc, 3, RB);
  { int n = 2048*64*26;   k_pool<64,54,26,4,2><<<(n+255)/256,256,0,stream>>>(RB, RA); }
  k_conv_wmma<64,5,64,64,26,26,2><<<2048,256,0,stream>>>(RA, wq[4], Bv[4], s, 4, wsc, 4, RB);
  { int n = 2048*64*13;   k_pool<64,26,13,2,2><<<(n+255)/256,256,0,stream>>>(RB, RA); }
  k_conv_wmma<64,3,64,64,13,13,1><<<2048,256,0,stream>>>(RA, wq[5], Bv[5], s, 5, wsc, 5, RB);
  { int n = 2048*64*6;    k_pool<64,13,6,2,2><<<(n+255)/256,256,0,stream>>>(RB, RA); }
  k_conv_wmma<64,3,72,80,6,6,1><<<2048,256,0,stream>>>(RA, wq[6], Bv[6], s, 6, wsc, 6, RB);
  { int n = 2048*72*3;    k_pool<72,6,3,2,2><<<(n+255)/256,256,0,stream>>>(RB, RA); }

  // flatten is a no-op on (B,72,3) contiguous -> (B,216)
  k_lin1_wmma<<<128, 128, 0, stream>>>(RA, wq[7], Bv[7], s, wsc, RB);
  { int n = 2048*17; k_lin2<<<(n+255)/256, 256, 0, stream>>>(RB, wq[8], Bv[8], s, wsc, outp); }
}
